// TRMAttention_66477503808163
// MI455X (gfx1250) — matrix operand order repaired and checked by reading, not compiled, not run on hardware
//
#include <hip/hip_runtime.h>
#include <hip/hip_bf16.h>

// ---------------------------------------------------------------------------
// Problem constants (match reference)
// ---------------------------------------------------------------------------
#define D_MODEL 2048
#define NUM_HEADS 16
#define HEAD_DIM 128
#define BATCH 2
#define SEQ 2048
#define MROWS (BATCH * SEQ)   // 4096 token rows

typedef __attribute__((ext_vector_type(16))) _Float16 v16h;
typedef __attribute__((ext_vector_type(8)))  float    v8f;
typedef __attribute__((ext_vector_type(4)))  int      v4i;

#if defined(__has_builtin)
#if __has_builtin(__builtin_amdgcn_global_load_async_to_lds_b128)
#define USE_ASYNC_LDS 1
#endif
#endif

// ---------------------------------------------------------------------------
// WMMA helpers (gfx1250: D = A[16x32 f16] * B[32x16 f16] + C[16x16 f32])
// A layout: lane L holds row M=L&15; halves 0..7 = K 8*(L>>4)..+7,
//           halves 8..15 = K 16+8*(L>>4)..+7
// B layout: lane L holds col N=L&15; halves 0..7 = K 8*(L>>4)..+7,
//           halves 8..15 = K 16+8*(L>>4)..+7 (strided gather)
// C layout: lane L holds col N=L&15; VGPR v holds row M=v+8*(L>>4)
// ---------------------------------------------------------------------------
__device__ __forceinline__ v8f wmma_f16(v16h a, v16h b, v8f c) {
    return __builtin_amdgcn_wmma_f32_16x16x32_f16(
        false, a, false, b, (short)0, c, false, false);
}

__device__ __forceinline__ v16h ld_afrag(const _Float16* row_base, int kb) {
    union { v16h v; uint4 q[2]; } u;
    u.q[0] = *reinterpret_cast<const uint4*>(row_base + kb);
    u.q[1] = *reinterpret_cast<const uint4*>(row_base + kb + 16);
    return u.v;
}

__device__ __forceinline__ v16h ld_bfrag(const _Float16* p, size_t ld) {
    v16h v;
#pragma unroll
    for (int s = 0; s < 8; ++s) {
        v[s]     = p[(size_t)s * ld];
        v[8 + s] = p[(size_t)(16 + s) * ld];
    }
    return v;
}

__device__ __forceinline__ void wait_async0() {
#if defined(USE_ASYNC_LDS)
#if __has_builtin(__builtin_amdgcn_s_wait_asynccnt)
    __builtin_amdgcn_s_wait_asynccnt(0);
#else
    asm volatile("s_wait_asynccnt 0" ::: "memory");
#endif
#endif
}

// ---------------------------------------------------------------------------
// f32 -> f16 conversion (grid-stride)
// ---------------------------------------------------------------------------
__global__ void convert_f16_kernel(const float* __restrict__ src,
                                   _Float16* __restrict__ dst, int n) {
    for (int i = blockIdx.x * blockDim.x + threadIdx.x; i < n;
         i += gridDim.x * blockDim.x)
        dst[i] = (_Float16)src[i];
}

// W [N,K] f32 row-major  ->  Wt [K,N] f16 (transposed) via LDS tile
__global__ void transpose_convert_kernel(const float* __restrict__ W,
                                         _Float16* __restrict__ Wt) {
    __shared__ float tile[32][33];
    const int k0 = blockIdx.x * 32;
    const int n0 = blockIdx.y * 32;
    for (int i = threadIdx.y; i < 32; i += 8)
        tile[i][threadIdx.x] = W[(size_t)(n0 + i) * D_MODEL + k0 + threadIdx.x];
    __syncthreads();
    for (int i = threadIdx.y; i < 32; i += 8)
        Wt[(size_t)(k0 + i) * D_MODEL + n0 + threadIdx.x] =
            (_Float16)tile[threadIdx.x][i];
}

// ---------------------------------------------------------------------------
// B-tile staging: 32(K) x 128(N) f16 panel into LDS, padded row stride 136
// halves (272 B) to spread strided 16-bit ld_bfrag reads across banks.
// Each of the 128 threads moves 4 x 16B chunks.
// ---------------------------------------------------------------------------
#define LDB 136

#if defined(USE_ASYNC_LDS)
__device__ __forceinline__ void stage_async(const _Float16* __restrict__ Wt,
                                            int n0, int k0,
                                            _Float16* __restrict__ dst, int tid) {
#pragma unroll
    for (int j = 0; j < 4; ++j) {
        const int cid = tid + 128 * j;        // 512 chunks of 16B
        const int k   = cid >> 4;             // 0..31
        const int col = cid & 15;             // 16B chunk within row
        const _Float16* g = Wt + (size_t)(k0 + k) * D_MODEL + n0 + col * 8;
        _Float16* l = dst + k * LDB + col * 8;
        __builtin_amdgcn_global_load_async_to_lds_b128(
            (v4i*)g, (v4i*)l, 0, 0);
    }
}
#else
__device__ __forceinline__ void stage_regs_load(uint4 r[4],
                                                const _Float16* __restrict__ Wt,
                                                int n0, int k0, int tid) {
#pragma unroll
    for (int j = 0; j < 4; ++j) {
        const int cid = tid + 128 * j;
        const int k   = cid >> 4;
        const int col = cid & 15;
        r[j] = *reinterpret_cast<const uint4*>(
            Wt + (size_t)(k0 + k) * D_MODEL + n0 + col * 8);
    }
}
__device__ __forceinline__ void stage_regs_store(const uint4 r[4],
                                                 _Float16* __restrict__ dst,
                                                 int tid) {
#pragma unroll
    for (int j = 0; j < 4; ++j) {
        const int cid = tid + 128 * j;
        const int k   = cid >> 4;
        const int col = cid & 15;
        *reinterpret_cast<uint4*>(dst + k * LDB + col * 8) = r[j];
    }
}
#endif

// ---------------------------------------------------------------------------
// Tiled GEMM:  C[M=4096, N=2048] = X[M,2048] @ Wt[2048,N]   (Wt = W^T)
// Block = 4 waves, tile 128M x 128N; wave = 32M x 128N (two 16-row subtiles).
// B panel staged in LDS (shared by all 4 waves), double-buffered with async
// global->LDS copies when available.
// MODE 0: Q -> RoPE -> Qbuf [B,H,S,hd] f16
// MODE 1: K -> RoPE -> KtT  [B,H,hd,S] f16 (transposed for attn B-operand)
// MODE 2: V ->         Vbuf [B,H,S,hd] f16
// MODE 3: out proj ->  d_out [B,S,D]   f32
// ---------------------------------------------------------------------------
template <int MODE>
__global__ __launch_bounds__(128)
void gemm_kernel(const _Float16* __restrict__ X,
                 const _Float16* __restrict__ Wt,
                 void* __restrict__ Out,
                 const float* __restrict__ cosT,
                 const float* __restrict__ sinT) {
    __shared__ __align__(16) _Float16 bs[2][32 * LDB];

    const int tid = threadIdx.x;
    const int L   = tid & 31;
    const int w   = tid >> 5;
    const int ln  = L & 15;
    const int hh  = L >> 4;
    const int m0  = blockIdx.y * 128 + w * 32;   // wave's token-row base
    const int n0  = blockIdx.x * 128;            // output-feature tile base

    v8f c[2][8] = {};
    const _Float16* xrow0 = X + (size_t)(m0 + ln) * D_MODEL;
    const _Float16* xrow1 = xrow0 + (size_t)16 * D_MODEL;

#if defined(USE_ASYNC_LDS)
    stage_async(Wt, n0, 0, &bs[0][0], tid);
    int cur = 0;
    for (int k0 = 0; k0 < D_MODEL; k0 += 32) {
        wait_async0();
        __syncthreads();                       // tile k0 visible to all waves
        if (k0 + 32 < D_MODEL)
            stage_async(Wt, n0, k0 + 32, &bs[cur ^ 1][0], tid);
        const _Float16* bp = &bs[cur][0];
        v16h a0 = ld_afrag(xrow0, k0 + 8 * hh);
        v16h a1 = ld_afrag(xrow1, k0 + 8 * hh);
#pragma unroll
        for (int f = 0; f < 8; ++f) {
            v16h b = ld_bfrag(bp + 8 * hh * LDB + ln + f * 16, LDB);
            c[0][f] = wmma_f16(a0, b, c[0][f]);
            c[1][f] = wmma_f16(a1, b, c[1][f]);
        }
        cur ^= 1;
    }
#else
    uint4 regs[4];
    stage_regs_load(regs, Wt, n0, 0, tid);
    for (int k0 = 0; k0 < D_MODEL; k0 += 32) {
        __syncthreads();                       // previous tile fully consumed
        stage_regs_store(regs, &bs[0][0], tid);
        if (k0 + 32 < D_MODEL)
            stage_regs_load(regs, Wt, n0, k0 + 32, tid);
        __syncthreads();                       // staged tile visible
        const _Float16* bp = &bs[0][0];
        v16h a0 = ld_afrag(xrow0, k0 + 8 * hh);
        v16h a1 = ld_afrag(xrow1, k0 + 8 * hh);
#pragma unroll
        for (int f = 0; f < 8; ++f) {
            v16h b = ld_bfrag(bp + 8 * hh * LDB + ln + f * 16, LDB);
            c[0][f] = wmma_f16(a0, b, c[0][f]);
            c[1][f] = wmma_f16(a1, b, c[1][f]);
        }
    }
#endif

    // Epilogue
#pragma unroll
    for (int mi = 0; mi < 2; ++mi) {
#pragma unroll
        for (int f = 0; f < 8; ++f) {
            const int col = n0 + 16 * f + ln;
            const int h   = col >> 7;
            const int off = col & 127;
#pragma unroll
            for (int v = 0; v < 8; ++v) {
                const int m = m0 + 16 * mi + v + 8 * hh;
                const int b = m >> 11;
                const int s = m & (SEQ - 1);
                float val = c[mi][f][v];
                if (MODE == 0 || MODE == 1) {
                    const float pair = c[mi][f ^ 4][v];
                    const float rot  = (f < 4) ? -pair : pair;
                    const float cv = cosT[s * HEAD_DIM + off];
                    const float sv = sinT[s * HEAD_DIM + off];
                    val = val * cv + rot * sv;
                }
                if (MODE == 0) {
                    ((_Float16*)Out)[(((size_t)(b * NUM_HEADS + h) * SEQ) + s) * HEAD_DIM + off] =
                        (_Float16)val;
                } else if (MODE == 1) {
                    ((_Float16*)Out)[(((size_t)(b * NUM_HEADS + h) * HEAD_DIM) + off) * SEQ + s] =
                        (_Float16)val;
                } else if (MODE == 2) {
                    ((_Float16*)Out)[(((size_t)(b * NUM_HEADS + h) * SEQ) + s) * HEAD_DIM + off] =
                        (_Float16)val;
                } else {
                    ((float*)Out)[(size_t)m * D_MODEL + col] = val;
                }
            }
        }
    }
}

// ---------------------------------------------------------------------------
// Flash attention: one wave per (b*H+h, 32-query tile). Online softmax in
// base-2 domain. Every K/V fragment feeds two WMMAs (two query subtiles).
// ---------------------------------------------------------------------------
#define LDP 48   // padded halves per P row (96B, 16B-aligned reads)

__global__ __launch_bounds__(32)
void attn_kernel(const _Float16* __restrict__ Q,
                 const _Float16* __restrict__ Kt,   // [B,H,hd,S]
                 const _Float16* __restrict__ V,    // [B,H,S,hd]
                 _Float16* __restrict__ O) {        // [B,S,D] f16
    __shared__ __align__(16) _Float16 plds[32 * LDP];

    const int bh = blockIdx.x;
    const int q0 = blockIdx.y * 32;
    const int L  = threadIdx.x;
    const int ln = L & 15;
    const int hh = L >> 4;

    const _Float16* Qb  = Q  + (size_t)bh * SEQ * HEAD_DIM;
    const _Float16* Ktb = Kt + (size_t)bh * HEAD_DIM * SEQ;
    const _Float16* Vb  = V  + (size_t)bh * SEQ * HEAD_DIM;

    // Q tiles: 2 x (16 queries x 128 dims) = 8 A-fragments
    v16h aq[2][4];
#pragma unroll
    for (int mi = 0; mi < 2; ++mi) {
        const _Float16* qrow = Qb + (size_t)(q0 + 16 * mi + ln) * HEAD_DIM;
#pragma unroll
        for (int kc = 0; kc < 4; ++kc)
            aq[mi][kc] = ld_afrag(qrow, 32 * kc + 8 * hh);
    }

    // fold 1/sqrt(hd) and log2(e) into one scale; softmax runs in exp2 domain
    const float scale = 0.08838834764831845f * 1.4426950408889634f;
    const float NEG   = -3.0e38f;

    float mrun[2][8], lrun[2][8];
    v8f acc[2][8] = {};
#pragma unroll
    for (int mi = 0; mi < 2; ++mi)
#pragma unroll
        for (int v = 0; v < 8; ++v) { mrun[mi][v] = NEG; lrun[mi][v] = 0.f; }

    for (int n0 = 0; n0 <= q0 + 31; n0 += 32) {
        // scores for both query subtiles over keys [n0, n0+32)
        v8f s[2][2] = {};
#pragma unroll
        for (int kc = 0; kc < 4; ++kc) {
            const _Float16* kb = Ktb + (size_t)(32 * kc + 8 * hh) * SEQ + n0 + ln;
            v16h b0 = ld_bfrag(kb, SEQ);
            v16h b1 = ld_bfrag(kb + 16, SEQ);
            s[0][0] = wmma_f16(aq[0][kc], b0, s[0][0]);
            s[1][0] = wmma_f16(aq[1][kc], b0, s[1][0]);
            s[0][1] = wmma_f16(aq[0][kc], b1, s[0][1]);
            s[1][1] = wmma_f16(aq[1][kc], b1, s[1][1]);
        }

        __syncthreads();   // previous iteration's ap reads are done
#pragma unroll
        for (int mi = 0; mi < 2; ++mi) {
            const int qbase = q0 + 16 * mi;
            float corr[8];
#pragma unroll
            for (int v = 0; v < 8; ++v) {
                const int qr = qbase + v + 8 * hh;
                float x0 = s[mi][0][v] * scale;
                float x1 = s[mi][1][v] * scale;
                if (n0 + ln      > qr) x0 = NEG;     // causal mask
                if (n0 + 16 + ln > qr) x1 = NEG;
                float mx = fmaxf(x0, x1);
                for (int d = 1; d < 16; d <<= 1)
                    mx = fmaxf(mx, __shfl_xor(mx, d, 32));
                const float mn = fmaxf(mrun[mi][v], mx);
                const float cr = exp2f(mrun[mi][v] - mn);
                const float p0 = exp2f(x0 - mn);
                const float p1 = exp2f(x1 - mn);
                float rs = p0 + p1;
                for (int d = 1; d < 16; d <<= 1)
                    rs += __shfl_xor(rs, d, 32);
                lrun[mi][v] = lrun[mi][v] * cr + rs;
                mrun[mi][v] = mn;
                corr[v] = cr;
                plds[(16 * mi + v + 8 * hh) * LDP + ln]      = (_Float16)p0;
                plds[(16 * mi + v + 8 * hh) * LDP + 16 + ln] = (_Float16)p1;
            }
#pragma unroll
            for (int f = 0; f < 8; ++f)
#pragma unroll
                for (int v = 0; v < 8; ++v)
                    acc[mi][f][v] *= corr[v];
        }
        __syncthreads();   // P tile staged

        v16h ap0 = ld_afrag(plds + (size_t)ln * LDP, 8 * hh);
        v16h ap1 = ld_afrag(plds + (size_t)(16 + ln) * LDP, 8 * hh);

        // acc += P @ V  (keys n0..n0+31); each V fragment used twice
#pragma unroll
        for (int f = 0; f < 8; ++f) {
            v16h vb = ld_bfrag(Vb + (size_t)(n0 + 8 * hh) * HEAD_DIM + 16 * f + ln, HEAD_DIM);
            acc[0][f] = wmma_f16(ap0, vb, acc[0][f]);
            acc[1][f] = wmma_f16(ap1, vb, acc[1][f]);
        }
    }

    // epilogue: normalize, scatter to token-major [B,S,D]
    const int b = bh >> 4;
    const int h = bh & 15;
#pragma unroll
    for (int mi = 0; mi < 2; ++mi) {
        float inv[8];
#pragma unroll
        for (int v = 0; v < 8; ++v) inv[v] = 1.f / lrun[mi][v];
#pragma unroll
        for (int f = 0; f < 8; ++f) {
            const int col = h * HEAD_DIM + 16 * f + ln;
#pragma unroll
            for (int v = 0; v < 8; ++v) {
                const int s = q0 + 16 * mi + v + 8 * hh;
                O[((size_t)(b * SEQ + s)) * D_MODEL + col] =
                    (_Float16)(acc[mi][f][v] * inv[v]);
            }
        }
    }
}

// ---------------------------------------------------------------------------
// Launch
// ---------------------------------------------------------------------------
extern "C" void kernel_launch(void* const* d_in, const int* in_sizes, int n_in,
                              void* d_out, int out_size, void* d_ws, size_t ws_size,
                              hipStream_t stream) {
    (void)in_sizes; (void)n_in; (void)out_size; (void)ws_size;
    const float* x    = (const float*)d_in[0];
    const float* cosT = (const float*)d_in[1];
    const float* sinT = (const float*)d_in[2];
    const float* Wq   = (const float*)d_in[3];
    const float* Wk   = (const float*)d_in[4];
    const float* Wv   = (const float*)d_in[5];
    const float* Wo   = (const float*)d_in[6];

    const size_t N_X = (size_t)MROWS * D_MODEL;
    const size_t N_W = (size_t)D_MODEL * D_MODEL;
    char* ws = (char*)d_ws;
    _Float16* xh   = (_Float16*)ws;                 ws += N_X * 2;
    _Float16* Wqt  = (_Float16*)ws;                 ws += N_W * 2;
    _Float16* Wkt  = (_Float16*)ws;                 ws += N_W * 2;
    _Float16* Wvt  = (_Float16*)ws;                 ws += N_W * 2;
    _Float16* Wot  = (_Float16*)ws;                 ws += N_W * 2;
    _Float16* Qbuf = (_Float16*)ws;                 ws += N_X * 2;
    _Float16* KtT  = (_Float16*)ws;                 ws += N_X * 2;
    _Float16* Vbuf = (_Float16*)ws;                 ws += N_X * 2;
    _Float16* Obuf = (_Float16*)ws;                 ws += N_X * 2;

    convert_f16_kernel<<<2048, 256, 0, stream>>>(x, xh, (int)N_X);
    dim3 tb(32, 8), tg(D_MODEL / 32, D_MODEL / 32);
    transpose_convert_kernel<<<tg, tb, 0, stream>>>(Wq, Wqt);
    transpose_convert_kernel<<<tg, tb, 0, stream>>>(Wk, Wkt);
    transpose_convert_kernel<<<tg, tb, 0, stream>>>(Wv, Wvt);
    transpose_convert_kernel<<<tg, tb, 0, stream>>>(Wo, Wot);

    dim3 gg(D_MODEL / 128, MROWS / 128);
    gemm_kernel<0><<<gg, 128, 0, stream>>>(xh, Wqt, Qbuf, cosT, sinT);
    gemm_kernel<1><<<gg, 128, 0, stream>>>(xh, Wkt, KtT,  cosT, sinT);
    gemm_kernel<2><<<gg, 128, 0, stream>>>(xh, Wvt, Vbuf, cosT, sinT);

    attn_kernel<<<dim3(BATCH * NUM_HEADS, SEQ / 32), 32, 0, stream>>>(
        Qbuf, KtT, Vbuf, Obuf);

    gemm_kernel<3><<<gg, 128, 0, stream>>>(Obuf, Wot, (float*)d_out, cosT, sinT);
}
